// ParallelGNNModel_64080912056903
// MI455X (gfx1250) — matrix operand order repaired and checked by reading, not compiled, not run on hardware
//
#include <hip/hip_runtime.h>
#include <hip/hip_bf16.h>
#include <math.h>

// ---------------------------------------------------------------------------
// Types for CDNA5 WMMA (wave32, 16x16x32 bf16 -> f32)
// ---------------------------------------------------------------------------
typedef __attribute__((ext_vector_type(16))) __bf16 v16bf;
typedef __attribute__((ext_vector_type(8)))  __bf16 v8bf;
typedef __attribute__((ext_vector_type(8)))  float  v8f;

union AFrag { v16bf v; v8bf h[2]; };

#define BN_SCALE 0.9999950000374997f  // 1/sqrt(1+1e-5), eval-mode BatchNorm1d
#define LN_EPS   1e-5f

__device__ __forceinline__ float gnn_lrelu(float v) { return v > 0.f ? v : 0.2f * v; }

__device__ __forceinline__ unsigned gnn_fkey(float f) {
  unsigned u = __float_as_uint(f);
  return (u & 0x80000000u) ? ~u : (u | 0x80000000u);
}
__device__ __forceinline__ float gnn_fdecode(unsigned u) {
  return __uint_as_float((u & 0x80000000u) ? (u ^ 0x80000000u) : ~u);
}

// ---------------------------------------------------------------------------
// Generic WMMA GEMM:  out[M, NT*16] = A[M,K] @ W[NT*16 rows of ldw]^T (+epilogue)
//   A: bf16 row-major [M, lda], K-padded to mult of 32
//   W: bf16 row-major [Nout, ldw] (row n holds weights for output col n)
// One wave computes a 16 x (NT*16) tile.  4 waves / block.
// EPI: 0 = (+bias) store; 1 = relu((acc+bias)*BN_SCALE); 2 = relu(LN(acc+bias))
// ---------------------------------------------------------------------------
template <int NT, int EPI>
__global__ __launch_bounds__(128) void gemm_wmma_kernel(
    const __bf16* __restrict__ A, int lda,
    const __bf16* __restrict__ W, int ldw,
    const float* __restrict__ bias,
    const float* __restrict__ lng, const float* __restrict__ lnb,
    float* __restrict__ outf, __bf16* __restrict__ outbf,
    int ldo, int M, int K, int ncol0) {
  const int lane = threadIdx.x & 31;
  const int wv   = threadIdx.x >> 5;
  const int mt   = blockIdx.x * 4 + wv;
  const int m0   = mt * 16;
  if (m0 >= M) return;
  const int cl   = lane & 15;
  const int half = lane >> 4;

  // A fragment: lane holds 16 contiguous K (matches B).
  const __bf16* arow = A + (size_t)(m0 + cl) * lda + half * 16;
  // B fragment: lane holds 16 contiguous K for output column (ncol0+nt*16+cl).
  const int boff = half * 16;

  v8f acc[NT] = {};
  for (int k0 = 0; k0 < K; k0 += 32) {
    AFrag a;
    a.h[0] = *(const v8bf*)(arow + k0);
    a.h[1] = *(const v8bf*)(arow + k0 + 8);
#pragma unroll
    for (int nt = 0; nt < NT; ++nt) {
      const __bf16* brow = W + (size_t)(ncol0 + nt * 16 + cl) * ldw + k0 + boff;
      v16bf b = *(const v16bf*)brow;
      acc[nt] = __builtin_amdgcn_wmma_f32_16x16x32_bf16(
          false, a.v, false, b, (short)0, acc[nt], false, false);
    }
  }

  if (EPI == 2) {
    // LayerNorm across the full (NT*16)-wide feature dim, then ReLU.
    float rs[8], rq[8];
#pragma unroll
    for (int r = 0; r < 8; ++r) { rs[r] = 0.f; rq[r] = 0.f; }
#pragma unroll
    for (int nt = 0; nt < NT; ++nt) {
      const float bc = bias[ncol0 + nt * 16 + cl];
#pragma unroll
      for (int r = 0; r < 8; ++r) {
        float v = acc[nt][r] + bc;
        acc[nt][r] = v;
        rs[r] += v; rq[r] += v * v;
      }
    }
#pragma unroll
    for (int r = 0; r < 8; ++r) {
#pragma unroll
      for (int m = 1; m < 16; m <<= 1) {  // reduce within the 16-lane half
        rs[r] += __shfl_xor(rs[r], m);
        rq[r] += __shfl_xor(rq[r], m);
      }
      const float invn = 1.0f / (float)(NT * 16);
      float mean = rs[r] * invn;
      float var  = rq[r] * invn - mean * mean;
      rs[r] = mean;
      rq[r] = rsqrtf(var + LN_EPS);
    }
#pragma unroll
    for (int nt = 0; nt < NT; ++nt) {
      const int col = ncol0 + nt * 16 + cl;
      const float g = lng[col], b2 = lnb[col];
#pragma unroll
      for (int r = 0; r < 8; ++r) {
        float v = (acc[nt][r] - rs[r]) * rq[r] * g + b2;
        v = fmaxf(v, 0.f);
        const size_t o = (size_t)(m0 + r + 8 * half) * ldo + col;
        if (outf)  outf[o]  = v;
        if (outbf) outbf[o] = (__bf16)v;
      }
    }
  } else {
#pragma unroll
    for (int nt = 0; nt < NT; ++nt) {
      const int col = ncol0 + nt * 16 + cl;
      const float bc = bias ? bias[col] : 0.f;
#pragma unroll
      for (int r = 0; r < 8; ++r) {
        float v = acc[nt][r] + bc;
        if (EPI == 1) v = fmaxf(v * BN_SCALE, 0.f);
        const size_t o = (size_t)(m0 + r + 8 * half) * ldo + col;
        if (outf)  outf[o]  = v;
        if (outbf) outbf[o] = (__bf16)v;
      }
    }
  }
}

// ---------------------------------------------------------------------------
// Utility / streaming kernels
// ---------------------------------------------------------------------------
__global__ void fill_f32_kernel(float* p, float v, size_t n) {
  size_t i = (size_t)blockIdx.x * blockDim.x + threadIdx.x;
  if (i < n) p[i] = v;
}
__global__ void fill_u32_kernel(unsigned* p, unsigned v, size_t n) {
  size_t i = (size_t)blockIdx.x * blockDim.x + threadIdx.x;
  if (i < n) p[i] = v;
}
// fp32 [R,C] -> bf16 [R,Cpad], zero-padded
__global__ void convert_pad_kernel(const float* __restrict__ src, __bf16* __restrict__ dst,
                                   int R, int C, int Cpad) {
  size_t i = (size_t)blockIdx.x * blockDim.x + threadIdx.x;
  if (i >= (size_t)R * Cpad) return;
  int r = (int)(i / Cpad), c = (int)(i % Cpad);
  dst[i] = (c < C) ? (__bf16)src[(size_t)r * C + c] : (__bf16)0.f;
}
__global__ void deg_kernel(const int* __restrict__ dst, float* deg, int E) {
  int e = blockIdx.x * blockDim.x + threadIdx.x;
  if (e < E) atomicAdd(deg + dst[e], 1.f);
}
__global__ void rsqrt_kernel(float* d, int n) {
  int i = blockIdx.x * blockDim.x + threadIdx.x;
  if (i < n) d[i] = rsqrtf(d[i]);
}
// y[i,c] = t[i,c] * dinv[i]^2   (self-loop contribution, also initializes y)
__global__ void selfloop_kernel(const float* __restrict__ t, const float* __restrict__ dinv,
                                float* __restrict__ y, size_t n) {
  size_t idx = (size_t)blockIdx.x * blockDim.x + threadIdx.x;
  if (idx >= n) return;
  int i = (int)(idx >> 7);
  float dv = dinv[i];
  y[idx] = t[idx] * dv * dv;
}
// y[d] += t[s] * dinv[s]*dinv[d]   (thread per (edge,col); 32-lane coalesced)
__global__ void gcn_scatter_kernel(const int* __restrict__ src, const int* __restrict__ dst,
                                   const float* __restrict__ dinv, const float* __restrict__ t,
                                   float* __restrict__ y, int E) {
  size_t idx = (size_t)blockIdx.x * blockDim.x + threadIdx.x;
  if (idx >= (size_t)E * 128) return;
  int e = (int)(idx >> 7), c = (int)(idx & 127);
  int s = src[e], d = dst[e];
  atomicAdd(y + (size_t)d * 128 + c, t[(size_t)s * 128 + c] * dinv[s] * dinv[d]);
}
// xg = relu((y + b)*BN_SCALE) + res  ; also writes bf16 mirror
__global__ void gcn_epi_kernel(const float* __restrict__ y, const float* __restrict__ res,
                               const float* __restrict__ b, float* __restrict__ xg,
                               __bf16* __restrict__ xgb, size_t n) {
  size_t idx = (size_t)blockIdx.x * blockDim.x + threadIdx.x;
  if (idx >= n) return;
  int c = (int)(idx & 127);
  float v = fmaxf((y[idx] + b[c]) * BN_SCALE, 0.f) + res[idx];
  xg[idx] = v;
  xgb[idx] = (__bf16)v;
}
// edge_attr column sums (block-local accumulate + shared-mem reduce)
__global__ void easum_kernel(const float* __restrict__ ea, float* __restrict__ sums, int E) {
  __shared__ float sh[256];
  float loc[10];
#pragma unroll
  for (int k = 0; k < 10; ++k) loc[k] = 0.f;
  for (int e = blockIdx.x * blockDim.x + threadIdx.x; e < E; e += gridDim.x * blockDim.x) {
#pragma unroll
    for (int k = 0; k < 10; ++k) loc[k] += ea[(size_t)e * 10 + k];
  }
  for (int k = 0; k < 10; ++k) {
    sh[threadIdx.x] = loc[k];
    __syncthreads();
    for (int s = 128; s > 0; s >>= 1) {
      if ((int)threadIdx.x < s) sh[threadIdx.x] += sh[threadIdx.x + s];
      __syncthreads();
    }
    if (threadIdx.x == 0) atomicAdd(sums + k, sh[0]);
    __syncthreads();
  }
}
// wemean[c] = (We @ ea_sum)[c] / E   (self-loop edge feature term)
__global__ void wemean_kernel(const float* __restrict__ We, const float* __restrict__ sums,
                              float* __restrict__ wm, float invE) {
  int c = threadIdx.x;  // 128 threads
  float s = 0.f;
#pragma unroll
  for (int k = 0; k < 10; ++k) s += We[c * 10 + k] * sums[k];
  wm[c] = s * invE;
}
// GATv2 pass A: logit_e = att . leaky_relu(hl[s]+hr[d]+We@ea_e); segment-max via u32 key
__global__ void gat_logits_kernel(const int* __restrict__ src, const int* __restrict__ dst,
                                  const float* __restrict__ ea, const float* __restrict__ We,
                                  const float* __restrict__ wemean,
                                  const float* __restrict__ hl, const float* __restrict__ hr,
                                  const float* __restrict__ att,
                                  float* __restrict__ logits, unsigned* __restrict__ mxkey,
                                  int E, int N) {
  int g = (int)(((size_t)blockIdx.x * blockDim.x + threadIdx.x) >> 5);
  int lane = threadIdx.x & 31;
  if (g >= E + N) return;
  int s, d;
  int c = lane * 4;
  float e0, e1, e2, e3;
  if (g < E) {
    s = src[g]; d = dst[g];
    float a[10];
#pragma unroll
    for (int k = 0; k < 10; ++k) a[k] = ea[(size_t)g * 10 + k];
    e0 = e1 = e2 = e3 = 0.f;
#pragma unroll
    for (int k = 0; k < 10; ++k) {
      e0 += a[k] * We[(c + 0) * 10 + k];
      e1 += a[k] * We[(c + 1) * 10 + k];
      e2 += a[k] * We[(c + 2) * 10 + k];
      e3 += a[k] * We[(c + 3) * 10 + k];
    }
  } else {
    s = d = g - E;
    e0 = wemean[c]; e1 = wemean[c + 1]; e2 = wemean[c + 2]; e3 = wemean[c + 3];
  }
  float4 l = *(const float4*)(hl + (size_t)s * 128 + c);
  float4 r = *(const float4*)(hr + (size_t)d * 128 + c);
  float m0 = gnn_lrelu(l.x + r.x + e0);
  float m1 = gnn_lrelu(l.y + r.y + e1);
  float m2 = gnn_lrelu(l.z + r.z + e2);
  float m3 = gnn_lrelu(l.w + r.w + e3);
  float p = m0 * att[c] + m1 * att[c + 1] + m2 * att[c + 2] + m3 * att[c + 3];
#pragma unroll
  for (int m = 16; m >= 1; m >>= 1) p += __shfl_xor(p, m);
  if (lane == 0) {
    logits[g] = p;
    atomicMax(mxkey + d, gnn_fkey(p));
  }
}
// GATv2 pass B: e = exp(logit - max[d]); z[d] += e
__global__ void gat_expsum_kernel(const int* __restrict__ dst, float* __restrict__ logits,
                                  const unsigned* __restrict__ mxkey, float* __restrict__ zsum,
                                  int E, int N) {
  int i = blockIdx.x * blockDim.x + threadIdx.x;
  if (i >= E + N) return;
  int d = (i < E) ? dst[i] : (i - E);
  float ev = __expf(logits[i] - gnn_fdecode(mxkey[d]));
  logits[i] = ev;
  atomicAdd(zsum + d, ev);
}
// GATv2 pass C: y[d] += hl[s] * alpha
__global__ void gat_scatter_kernel(const int* __restrict__ src, const int* __restrict__ dst,
                                   const float* __restrict__ logits, const float* __restrict__ zsum,
                                   const float* __restrict__ hl, float* __restrict__ y,
                                   int E, int N) {
  size_t idx = (size_t)blockIdx.x * blockDim.x + threadIdx.x;
  if (idx >= (size_t)(E + N) * 128) return;
  int i = (int)(idx >> 7), c = (int)(idx & 127);
  int s, d;
  if (i < E) { s = src[i]; d = dst[i]; } else { s = d = i - E; }
  float alpha = logits[i] / (zsum[d] + 1e-16f);
  atomicAdd(y + (size_t)d * 128 + c, hl[(size_t)s * 128 + c] * alpha);
}
// GAT epilogue fused with graph pooling into comb[:,128:256]
__global__ void gat_pool_kernel(const float* __restrict__ y, const float* __restrict__ gres,
                                const float* __restrict__ bias, const int* __restrict__ batch,
                                float* __restrict__ comb, size_t n) {
  size_t idx = (size_t)blockIdx.x * blockDim.x + threadIdx.x;
  if (idx >= n) return;
  int i = (int)(idx >> 7), c = (int)(idx & 127);
  float v = fmaxf((y[idx] + bias[c]) * BN_SCALE, 0.f) + gres[idx];
  atomicAdd(comb + (size_t)batch[i] * 256 + 128 + c, v);
}
// GCN graph pooling into comb[:,0:128]
__global__ void gcn_pool_kernel(const float* __restrict__ xg, const int* __restrict__ batch,
                                float* __restrict__ comb, size_t n) {
  size_t idx = (size_t)blockIdx.x * blockDim.x + threadIdx.x;
  if (idx >= n) return;
  int i = (int)(idx >> 7), c = (int)(idx & 127);
  atomicAdd(comb + (size_t)batch[i] * 256 + c, xg[idx]);
}
// Final head: out[g] = z[g,:] . c3_W + c3_b
__global__ void c3_kernel(const float* __restrict__ z, const float* __restrict__ Wc,
                          const float* __restrict__ bc, float* __restrict__ out, int G) {
  int i = blockIdx.x * blockDim.x + threadIdx.x;
  if (i >= G) return;
  float s = bc[0];
#pragma unroll
  for (int k = 0; k < 64; ++k) s += z[(size_t)i * 64 + k] * Wc[k];
  out[i] = s;
}

// ---------------------------------------------------------------------------
// Host orchestration
// ---------------------------------------------------------------------------
static inline unsigned gcdiv(size_t n, size_t b) { return (unsigned)((n + b - 1) / b); }

extern "C" void kernel_launch(void* const* d_in, const int* in_sizes, int n_in,
                              void* d_out, int out_size, void* d_ws, size_t ws_size,
                              hipStream_t stream) {
  (void)n_in; (void)out_size; (void)ws_size;
  const float* x      = (const float*)d_in[0];
  const int*   ei     = (const int*)d_in[1];
  const float* ea     = (const float*)d_in[2];
  const int*   batch  = (const int*)d_in[3];
  const float* gfe    = (const float*)d_in[4];
  const float* W0     = (const float*)d_in[5];
  const float* b0     = (const float*)d_in[6];
  const float* ln0g   = (const float*)d_in[7];
  const float* ln0b   = (const float*)d_in[8];
  const float* gcnW   = (const float*)d_in[9];
  const float* gcnb   = (const float*)d_in[10];
  const float* resW   = (const float*)d_in[11];
  const float* resb   = (const float*)d_in[12];
  const float* Wl     = (const float*)d_in[13];
  const float* bl     = (const float*)d_in[14];
  const float* Wr     = (const float*)d_in[15];
  const float* br     = (const float*)d_in[16];
  const float* We     = (const float*)d_in[17];
  const float* att    = (const float*)d_in[18];
  const float* gbias  = (const float*)d_in[19];
  const float* rgW    = (const float*)d_in[20];
  const float* rgb    = (const float*)d_in[21];
  const float* g1W    = (const float*)d_in[22];
  const float* g1b    = (const float*)d_in[23];
  const float* g1g    = (const float*)d_in[24];
  const float* g1bb   = (const float*)d_in[25];
  const float* g2W    = (const float*)d_in[26];
  const float* g2b    = (const float*)d_in[27];
  const float* g2g    = (const float*)d_in[28];
  const float* g2bb   = (const float*)d_in[29];
  const float* f1W    = (const float*)d_in[30];
  const float* f1b    = (const float*)d_in[31];
  const float* f2W    = (const float*)d_in[32];
  const float* f2b    = (const float*)d_in[33];
  const float* c1W    = (const float*)d_in[34];
  const float* c1b    = (const float*)d_in[35];
  const float* c2W    = (const float*)d_in[36];
  const float* c2b    = (const float*)d_in[37];
  const float* c3W    = (const float*)d_in[38];
  const float* c3b    = (const float*)d_in[39];
  float* out = (float*)d_out;

  const int N = in_sizes[0] / 74;
  const int E = in_sizes[1] / 2;
  const int G = in_sizes[4] / 8;
  const int* src = ei;
  const int* dst = ei + E;

  // -------- workspace layout (256B aligned slabs) --------
  char* wsbase = (char*)d_ws;
  size_t off = 0;
  auto ALLOC = [&](size_t bytes) -> char* {
    char* p = wsbase + off;
    off = (off + bytes + 255) & ~(size_t)255;
    return p;
  };
  float*    dinv    = (float*)ALLOC((size_t)N * 4);
  __bf16*   x_bf    = (__bf16*)ALLOC((size_t)N * 96 * 2);
  __bf16*   h_bf    = (__bf16*)ALLOC((size_t)N * 128 * 2);
  float*    xg      = (float*)ALLOC((size_t)N * 128 * 4);
  __bf16*   xg_bf   = (__bf16*)ALLOC((size_t)N * 128 * 2);
  float*    tbuf    = (float*)ALLOC((size_t)N * 128 * 4);   // GCN h@W  | GAT hl
  float*    rbuf    = (float*)ALLOC((size_t)N * 128 * 4);   // GCN res  | GAT hr
  float*    ybuf    = (float*)ALLOC((size_t)N * 128 * 4);   // GCN agg  | GAT agg
  float*    gres    = (float*)ALLOC((size_t)N * 128 * 4);
  float*    logits  = (float*)ALLOC((size_t)(E + N) * 4);
  unsigned* mxkey   = (unsigned*)ALLOC((size_t)N * 4);
  float*    zsum    = (float*)ALLOC((size_t)N * 4);
  float*    easum   = (float*)ALLOC(64);
  float*    wemean  = (float*)ALLOC(512);
  float*    comb    = (float*)ALLOC((size_t)G * 256 * 4);
  __bf16*   comb_bf = (__bf16*)ALLOC((size_t)G * 256 * 2);
  __bf16*   f_bf    = (__bf16*)ALLOC((size_t)G * 256 * 2);
  __bf16*   z_bf    = (__bf16*)ALLOC((size_t)G * 192 * 2);
  __bf16*   zc1_bf  = (__bf16*)ALLOC((size_t)G * 128 * 2);
  float*    zc2     = (float*)ALLOC((size_t)G * 64 * 4);
  __bf16*   gfp_bf  = (__bf16*)ALLOC((size_t)G * 32 * 2);
  __bf16*   gf1_bf  = (__bf16*)ALLOC((size_t)G * 128 * 2);
  __bf16*   w0b     = (__bf16*)ALLOC((size_t)128 * 96 * 2);
  __bf16*   gcnWb   = (__bf16*)ALLOC((size_t)3 * 128 * 128 * 2);
  __bf16*   resWb   = (__bf16*)ALLOC((size_t)3 * 128 * 128 * 2);
  __bf16*   wlb     = (__bf16*)ALLOC((size_t)128 * 128 * 2);
  __bf16*   wrb     = (__bf16*)ALLOC((size_t)128 * 128 * 2);
  __bf16*   rgatb   = (__bf16*)ALLOC((size_t)128 * 128 * 2);
  __bf16*   g1wb    = (__bf16*)ALLOC((size_t)128 * 32 * 2);
  __bf16*   g2wb    = (__bf16*)ALLOC((size_t)64 * 128 * 2);
  __bf16*   f1wb    = (__bf16*)ALLOC((size_t)256 * 256 * 2);
  __bf16*   f2wb    = (__bf16*)ALLOC((size_t)128 * 256 * 2);
  __bf16*   c1wb    = (__bf16*)ALLOC((size_t)128 * 192 * 2);
  __bf16*   c2wb    = (__bf16*)ALLOC((size_t)64 * 128 * 2);

  const unsigned BLK = 256;
  auto CVT = [&](const float* s, __bf16* dd, int R, int C, int Cp) {
    convert_pad_kernel<<<gcdiv((size_t)R * Cp, BLK), BLK, 0, stream>>>(s, dd, R, C, Cp);
  };

  // -------- bf16 conversions (activations K-padded to mult of 32) --------
  CVT(x, x_bf, N, 74, 96);
  CVT(W0, w0b, 128, 74, 96);
  CVT(gcnW, gcnWb, 3 * 128, 128, 128);
  CVT(resW, resWb, 3 * 128, 128, 128);
  CVT(Wl, wlb, 128, 128, 128);
  CVT(Wr, wrb, 128, 128, 128);
  CVT(rgW, rgatb, 128, 128, 128);
  CVT(g1W, g1wb, 128, 8, 32);
  CVT(g2W, g2wb, 64, 128, 128);
  CVT(f1W, f1wb, 256, 256, 256);
  CVT(f2W, f2wb, 128, 256, 256);
  CVT(c1W, c1wb, 128, 192, 192);
  CVT(c2W, c2wb, 64, 128, 128);
  CVT(gfe, gfp_bf, G, 8, 32);

  // -------- symmetric GCN normalization: dinv = rsqrt(1 + indegree) --------
  fill_f32_kernel<<<gcdiv(N, BLK), BLK, 0, stream>>>(dinv, 1.0f, (size_t)N);
  deg_kernel<<<gcdiv(E, BLK), BLK, 0, stream>>>(dst, dinv, E);
  rsqrt_kernel<<<gcdiv(N, BLK), BLK, 0, stream>>>(dinv, N);

  auto GEMM_GRID = [&](int M) { return dim3((unsigned)(((M / 16) + 3) / 4)); };

  // -------- layer0: h = relu(LN(x @ W0^T + b0)) --------
  gemm_wmma_kernel<8, 2><<<GEMM_GRID(N), 128, 0, stream>>>(
      x_bf, 96, w0b, 96, b0, ln0g, ln0b, nullptr, h_bf, 128, N, 96, 0);

  // -------- 3 GCN layers --------
  const __bf16* curb = h_bf;
  for (int i = 0; i < 3; ++i) {
    gemm_wmma_kernel<8, 0><<<GEMM_GRID(N), 128, 0, stream>>>(
        curb, 128, gcnWb + (size_t)i * 128 * 128, 128, nullptr, nullptr, nullptr,
        tbuf, nullptr, 128, N, 128, 0);
    gemm_wmma_kernel<8, 0><<<GEMM_GRID(N), 128, 0, stream>>>(
        curb, 128, resWb + (size_t)i * 128 * 128, 128, resb + i * 128, nullptr, nullptr,
        rbuf, nullptr, 128, N, 128, 0);
    selfloop_kernel<<<gcdiv((size_t)N * 128, BLK), BLK, 0, stream>>>(tbuf, dinv, ybuf, (size_t)N * 128);
    gcn_scatter_kernel<<<gcdiv((size_t)E * 128, BLK), BLK, 0, stream>>>(src, dst, dinv, tbuf, ybuf, E);
    gcn_epi_kernel<<<gcdiv((size_t)N * 128, BLK), BLK, 0, stream>>>(
        ybuf, rbuf, gcnb + i * 128, xg, xg_bf, (size_t)N * 128);
    curb = xg_bf;
  }

  // -------- GATv2 branch (reuses tbuf=hl, rbuf=hr, ybuf=agg) --------
  gemm_wmma_kernel<8, 0><<<GEMM_GRID(N), 128, 0, stream>>>(
      h_bf, 128, wlb, 128, bl, nullptr, nullptr, tbuf, nullptr, 128, N, 128, 0);
  gemm_wmma_kernel<8, 0><<<GEMM_GRID(N), 128, 0, stream>>>(
      h_bf, 128, wrb, 128, br, nullptr, nullptr, rbuf, nullptr, 128, N, 128, 0);
  gemm_wmma_kernel<8, 0><<<GEMM_GRID(N), 128, 0, stream>>>(
      h_bf, 128, rgatb, 128, rgb, nullptr, nullptr, gres, nullptr, 128, N, 128, 0);

  fill_f32_kernel<<<1, 32, 0, stream>>>(easum, 0.f, 16);
  easum_kernel<<<256, 256, 0, stream>>>(ea, easum, E);
  wemean_kernel<<<1, 128, 0, stream>>>(We, easum, wemean, 1.0f / (float)E);

  fill_u32_kernel<<<gcdiv(N, BLK), BLK, 0, stream>>>(mxkey, 0x007FFFFFu /* key(-inf) */, (size_t)N);
  fill_f32_kernel<<<gcdiv(N, BLK), BLK, 0, stream>>>(zsum, 0.f, (size_t)N);
  fill_f32_kernel<<<gcdiv((size_t)N * 128, BLK), BLK, 0, stream>>>(ybuf, 0.f, (size_t)N * 128);

  gat_logits_kernel<<<gcdiv((size_t)(E + N) * 32, 128), 128, 0, stream>>>(
      src, dst, ea, We, wemean, tbuf, rbuf, att, logits, mxkey, E, N);
  gat_expsum_kernel<<<gcdiv(E + N, BLK), BLK, 0, stream>>>(dst, logits, mxkey, zsum, E, N);
  gat_scatter_kernel<<<gcdiv((size_t)(E + N) * 128, BLK), BLK, 0, stream>>>(
      src, dst, logits, zsum, tbuf, ybuf, E, N);

  // -------- pooling into comb = [gcn_pool | gat_pool] --------
  fill_f32_kernel<<<gcdiv((size_t)G * 256, BLK), BLK, 0, stream>>>(comb, 0.f, (size_t)G * 256);
  gcn_pool_kernel<<<gcdiv((size_t)N * 128, BLK), BLK, 0, stream>>>(xg, batch, comb, (size_t)N * 128);
  gat_pool_kernel<<<gcdiv((size_t)N * 128, BLK), BLK, 0, stream>>>(
      ybuf, gres, gbias, batch, comb, (size_t)N * 128);
  CVT(comb, comb_bf, G, 256, 256);

  // -------- head MLPs --------
  gemm_wmma_kernel<8, 1><<<GEMM_GRID(G), 128, 0, stream>>>(
      comb_bf, 256, f1wb, 256, f1b, nullptr, nullptr, nullptr, f_bf, 256, G, 256, 0);
  gemm_wmma_kernel<8, 1><<<GEMM_GRID(G), 128, 0, stream>>>(
      comb_bf, 256, f1wb, 256, f1b, nullptr, nullptr, nullptr, f_bf, 256, G, 256, 128);
  gemm_wmma_kernel<8, 1><<<GEMM_GRID(G), 128, 0, stream>>>(
      f_bf, 256, f2wb, 256, f2b, nullptr, nullptr, nullptr, z_bf, 192, G, 256, 0);
  gemm_wmma_kernel<8, 2><<<GEMM_GRID(G), 128, 0, stream>>>(
      gfp_bf, 32, g1wb, 32, g1b, g1g, g1bb, nullptr, gf1_bf, 128, G, 32, 0);
  gemm_wmma_kernel<4, 2><<<GEMM_GRID(G), 128, 0, stream>>>(
      gf1_bf, 128, g2wb, 128, g2b, g2g, g2bb, nullptr, z_bf + 128, 192, G, 128, 0);
  gemm_wmma_kernel<8, 1><<<GEMM_GRID(G), 128, 0, stream>>>(
      z_bf, 192, c1wb, 192, c1b, nullptr, nullptr, nullptr, zc1_bf, 128, G, 192, 0);
  gemm_wmma_kernel<4, 1><<<GEMM_GRID(G), 128, 0, stream>>>(
      zc1_bf, 128, c2wb, 128, c2b, nullptr, nullptr, zc2, nullptr, 64, G, 128, 0);
  c3_kernel<<<gcdiv(G, 64), 64, 0, stream>>>(zc2, c3W, c3b, out, G);
}
